// FP8LinearFast_5832565588649
// MI455X (gfx1250) — 8-bit matrix operand order repaired and checked by reading, compiled, not run on hardware
//
#include <hip/hip_runtime.h>
#include <hip/hip_fp16.h>
#include <hip/hip_bf16.h>
#include <stdint.h>

#define FP8_E5M2_MAX 57344.0f

typedef int   v16i __attribute__((ext_vector_type(16)));
typedef float v8f  __attribute__((ext_vector_type(8)));

#if __has_builtin(__builtin_amdgcn_wmma_f32_16x16x128_bf8_fp8)
#define WMMA_BF8FP8 __builtin_amdgcn_wmma_f32_16x16x128_bf8_fp8
#else
#define WMMA_BF8FP8 __builtin_amdgcn_wmma_f32_16x16x128_fp8_fp8
#endif


__device__ __forceinline__ uint32_t f32_to_e5m2(float x) {
    x = fminf(fmaxf(x, -FP8_E5M2_MAX), FP8_E5M2_MAX);
    unsigned short hb = __half_as_ushort(__float2half_rn(x));
    uint32_t sign = (hb >> 8) & 0x80u;
    uint32_t mag  = hb & 0x7FFFu;
    uint32_t lsb  = (mag >> 8) & 1u;
    mag = mag + 0x7Fu + lsb;
    uint32_t b = mag >> 8;
    if (b > 0x7Bu) b = 0x7Bu;
    return sign | b;
}

__device__ __forceinline__ uint32_t f32_to_e4m3(float x) {
    uint32_t u = __float_as_uint(x);
    uint32_t sign = (u >> 24) & 0x80u;
    uint32_t a = u & 0x7FFFFFFFu;
    if (a >= 0x43E00000u) return sign | 0x7Eu;
    if (a < 0x3C800000u) {
        float m = __uint_as_float(a) * 512.0f;
        uint32_t q = (uint32_t)(m + 0.5f);
        if (q > 7u) q = 7u;
        return sign | q;
    }
    uint32_t lsb = (a >> 20) & 1u;
    a += 0x0007FFFFu + lsb;
    uint32_t exp  = (a >> 23) - 120u;
    uint32_t mant = (a >> 20) & 7u;
    return sign | (exp << 3) | mant;
}


__global__ __launch_bounds__(256) void k_absmax_part(const float4* __restrict__ x,
                                                     float* __restrict__ part,
                                                     long long nvec) {
    __shared__ float red[256];
    const int tid = threadIdx.x;
    long long i = (long long)blockIdx.x * 256 + tid;
    const long long stride = (long long)gridDim.x * 256;
    float m = 0.0f;
    for (; i < nvec; i += stride) {
        float4 v = x[i];
        m = fmaxf(m, fmaxf(fmaxf(fabsf(v.x), fabsf(v.y)),
                           fmaxf(fabsf(v.z), fabsf(v.w))));
    }
    red[tid] = m;
    __syncthreads();
    for (int s = 128; s > 0; s >>= 1) {
        if (tid < s) red[tid] = fmaxf(red[tid], red[tid + s]);
        __syncthreads();
    }
    if (tid == 0) part[blockIdx.x] = red[0];
}

__global__ __launch_bounds__(256) void k_absmax_fin(const float* __restrict__ part,
                                                    int n, float* __restrict__ scl) {
    __shared__ float red[256];
    const int tid = threadIdx.x;
    float m = 0.0f;
    for (int i = tid; i < n; i += 256) m = fmaxf(m, part[i]);
    red[tid] = m;
    __syncthreads();
    for (int s = 128; s > 0; s >>= 1) {
        if (tid < s) red[tid] = fmaxf(red[tid], red[tid + s]);
        __syncthreads();
    }
    if (tid == 0) {
        float s = red[0] / FP8_E5M2_MAX + 1e-12f;
        scl[0] = s;
        scl[1] = 1.0f / s;
    }
}

__global__ __launch_bounds__(256) void k_quant_x(const float* __restrict__ x,
                                                 const float* __restrict__ scl,
                                                 uint32_t* __restrict__ xq,
                                                 int K, int KC, long long ndw) {
    long long d = (long long)blockIdx.x * 256 + threadIdx.x;
    if (d >= ndw) return;
    const float inv = scl[1];
    const int v    = (int)(d & 15);
    const int lane = (int)((d >> 4) & 31);
    const long long tile = d >> 9;
    const int kc = (int)(tile % KC);
    const long long mtile = tile / KC;
    const int grp = lane >> 4;
    const int vl  = v & 7;
    const int k = kc * 128 + 64 * (v >> 3) + 16 * (vl >> 1) + 4 * (vl & 1) + 8 * grp;
    const long long row = mtile * 16 + (lane & 15);
    const float4 xv = *(const float4*)(x + (size_t)row * K + k);
    uint32_t b0 = f32_to_e5m2(xv.x * inv);
    uint32_t b1 = f32_to_e5m2(xv.y * inv);
    uint32_t b2 = f32_to_e5m2(xv.z * inv);
    uint32_t b3 = f32_to_e5m2(xv.w * inv);
    xq[d] = b0 | (b1 << 8) | (b2 << 16) | (b3 << 24);
}

__global__ __launch_bounds__(256) void k_quant_w(const float* __restrict__ w,
                                                 uint32_t* __restrict__ wq,
                                                 int K, int KC, long long ndw) {
    long long d = (long long)blockIdx.x * 256 + threadIdx.x;
    if (d >= ndw) return;
    const int v    = (int)(d & 15);
    const int lane = (int)((d >> 4) & 31);
    const long long tile = d >> 9;
    const int kc = (int)(tile % KC);
    const long long ntile = tile / KC;
    const int grp = lane >> 4;
    const int k = kc * 128 + 64 * (v >> 3) + 16 * ((v & 7) >> 1) + 4 * (v & 1) + 8 * grp;
    const long long n = ntile * 16 + (lane & 15);
    const float4 wv = *(const float4*)(w + (size_t)n * K + k);
    uint32_t b0 = f32_to_e4m3(wv.x);
    uint32_t b1 = f32_to_e4m3(wv.y);
    uint32_t b2 = f32_to_e4m3(wv.z);
    uint32_t b3 = f32_to_e4m3(wv.w);
    wq[d] = b0 | (b1 << 8) | (b2 << 16) | (b3 << 24);
}


__device__ __forceinline__ void issue_stage(const uint8_t* __restrict__ aq,
                                            const uint8_t* __restrict__ wq,
                                            uint8_t* lds, int mtB, int ntB,
                                            int KC, int kc, int buf, int tid) {
#pragma unroll
    for (int f = 0; f < 8; ++f) {
        const uint8_t* ga = aq + ((size_t)(mtB + f) * KC + kc) * 2048 + (size_t)tid * 16;
        uint32_t la = (uint32_t)(uintptr_t)(lds + buf * 32768 + f * 2048 + tid * 16);
        asm volatile("global_load_async_to_lds_b128 %0, %1, off"
                     :: "v"(la), "v"(ga) : "memory");
    }
#pragma unroll
    for (int f = 0; f < 8; ++f) {
        const uint8_t* ga = wq + ((size_t)(ntB + f) * KC + kc) * 2048 + (size_t)tid * 16;
        uint32_t la = (uint32_t)(uintptr_t)(lds + buf * 32768 + 16384 + f * 2048 + tid * 16);
        asm volatile("global_load_async_to_lds_b128 %0, %1, off"
                     :: "v"(la), "v"(ga) : "memory");
    }
}

__global__ __launch_bounds__(128) void k_gemm(const uint8_t* __restrict__ aq,
                                              const uint8_t* __restrict__ wq,
                                              const float* __restrict__ scl,
                                              const float* __restrict__ scale_w,
                                              const __hip_bfloat16* __restrict__ bias,
                                              __hip_bfloat16* __restrict__ out,
                                              int KC, int Nout) {
    extern __shared__ uint8_t lds[];
    const int tid   = threadIdx.x;
    const int lane  = tid & 31;
    const int wid   = tid >> 5;
    const int waveM = wid >> 1;
    const int waveN = wid & 1;
    const int mtB = blockIdx.y * 8;
    const int ntB = blockIdx.x * 8;

    union Frag { v16i v; int4 q[4]; };
    v8f acc[4][4] = {};

    issue_stage(aq, wq, lds, mtB, ntB, KC, 0, 0, tid);

    for (int kc = 0; kc < KC; ++kc) {
        const int buf = kc & 1;
        if (kc + 1 < KC) {
            issue_stage(aq, wq, lds, mtB, ntB, KC, kc + 1, buf ^ 1, tid);
            asm volatile("s_wait_asynccnt 0x10" ::: "memory");
        } else {
            asm volatile("s_wait_asynccnt 0x0" ::: "memory");
        }
        __syncthreads();

        Frag A[4], B[4];
#pragma unroll
        for (int i = 0; i < 4; ++i) {
            const int4* p = (const int4*)(lds + buf * 32768 +
                                          (waveM * 4 + i) * 2048 + lane * 64);
            A[i].q[0] = p[0]; A[i].q[1] = p[1]; A[i].q[2] = p[2]; A[i].q[3] = p[3];
        }
#pragma unroll
        for (int j = 0; j < 4; ++j) {
            const int4* p = (const int4*)(lds + buf * 32768 + 16384 +
                                          (waveN * 4 + j) * 2048 + lane * 64);
            B[j].q[0] = p[0]; B[j].q[1] = p[1]; B[j].q[2] = p[2]; B[j].q[3] = p[3];
        }
#pragma unroll
        for (int i = 0; i < 4; ++i)
#pragma unroll
            for (int j = 0; j < 4; ++j)
                acc[i][j] = WMMA_BF8FP8(A[i].v, B[j].v, (short)0, acc[i][j],
                                        false, false);

        __syncthreads();
    }

    const float sx       = scl[0];
    const int   colLocal = lane & 15;
    const int   rowHalf  = (lane >> 4) * 8;
#pragma unroll
    for (int j = 0; j < 4; ++j) {
        const int col = (ntB + waveN * 4 + j) * 16 + colLocal;
        const float cs = scale_w[col] * sx;
        const float bf = __bfloat162float(bias[col]);
#pragma unroll
        for (int i = 0; i < 4; ++i) {
            const int rowBase = (mtB + waveM * 4 + i) * 16 + rowHalf;
#pragma unroll
            for (int r = 0; r < 8; ++r) {
                float y = acc[i][j][r] * cs;
                __hip_bfloat16 t = __float2bfloat16(y);
                float o = __bfloat162float(t) + bf;
                out[(size_t)(rowBase + r) * Nout + col] = __float2bfloat16(o);
            }
        }
    }
}


extern "C" void kernel_launch(void* const* d_in, const int* in_sizes, int n_in,
                              void* d_out, int out_size, void* d_ws, size_t ws_size,
                              hipStream_t stream) {
    const float*          x       = (const float*)d_in[0];
    const float*          w       = (const float*)d_in[1];
    const float*          scale_w = (const float*)d_in[2];
    const __hip_bfloat16* bias    = (const __hip_bfloat16*)d_in[3];
    __hip_bfloat16*       out     = (__hip_bfloat16*)d_out;

    const long long OUT = in_sizes[2];
    const long long IN  = in_sizes[1] / OUT;
    const long long T   = (long long)in_sizes[0] / IN;
    const int K  = (int)IN;
    const int KC = K / 128;

    char*    ws   = (char*)d_ws;
    float*   part = (float*)ws;
    float*   scl  = (float*)(ws + 16384);
    uint8_t* xq   = (uint8_t*)(ws + 32768);
    uint8_t* wq   = xq + (size_t)T * K;

    const int ABS_BLOCKS = 4096;
    const long long nvec = (T * (long long)K) / 4;
    k_absmax_part<<<ABS_BLOCKS, 256, 0, stream>>>((const float4*)x, part, nvec);
    k_absmax_fin<<<1, 256, 0, stream>>>(part, ABS_BLOCKS, scl);

    const long long ndwx = (T * (long long)K) / 4;
    k_quant_x<<<(unsigned)((ndwx + 255) / 256), 256, 0, stream>>>(
        x, scl, (uint32_t*)xq, K, KC, ndwx);

    const long long ndww = (OUT * (long long)K) / 4;
    k_quant_w<<<(unsigned)((ndww + 255) / 256), 256, 0, stream>>>(
        w, (uint32_t*)wq, K, KC, ndww);

    dim3 grid((unsigned)(OUT / 128), (unsigned)(T / 128));
    k_gemm<<<grid, 128, 65536, stream>>>(xq, wq, scl, scale_w, bias, out, KC, (int)OUT);
}
